// MyModel_87522843558546
// MI455X (gfx1250) — matrix operand order repaired and checked by reading, not compiled, not run on hardware
//
#include <hip/hip_runtime.h>
#include <hip/hip_bf16.h>

// ---------------------------------------------------------------------------
// 2-layer LSTM, persistent-kernel design for MI455X (gfx1250, wave32, WMMA).
//
//  * Per layer: one persistent kernel, grid = 16 (H column groups) x 8 (batch
//    row groups) workgroups of 256 threads (8 waves).
//  * Each WG keeps its 1024(K) x 128(N) bf16 weight slice resident in LDS
//    (256 KB) for all 512 timesteps.
//  * A-tile split: x-half (K 0..511) is ASYNC-prefetched one step ahead into
//    a double-buffered LDS region with GLOBAL_LOAD_ASYNC_TO_LDS_B128;
//    h-half (K 512..1023) is async-loaded right after the inter-WG barrier.
//    One s_wait_asynccnt 4 per step waits for this step's operands while the
//    next-step prefetch stays in flight.
//  * Inner loop: 32 x v_wmma_f32_16x16x32_bf16 per wave per step over two
//    independent accumulators (hides XDL RAW latency).
//  * Cell state c lives in registers; h ping-pongs in workspace as bf16.
//  * Inter-WG sync per step: atomic-counter barrier among the 16 WGs of a
//    row group (counters zeroed via hipMemsetAsync each call -> graph-safe).
// ---------------------------------------------------------------------------

typedef __attribute__((ext_vector_type(16))) __bf16 v16bf;
typedef __attribute__((ext_vector_type(8)))  __bf16 v8bf;
typedef __attribute__((ext_vector_type(8)))  float  v8f;

#define Bsz   128
#define Tsz   512
#define Hsz   512
#define NWGN  16                // H column groups (32 h-cols each, x4 gates)
#define NWGM  8                 // batch row groups (16 rows each)

#define LDS_W_BYTES   (128 * 1024 * 2)   // 256 KB weight slice
#define AX_BYTES      (16 * 512 * 2)     // 16 KB x-half tile
#define AH_BYTES      (16 * 512 * 2)     // 16 KB h-half tile
#define OFF_AX0       LDS_W_BYTES
#define OFF_AX1       (OFF_AX0 + AX_BYTES)
#define OFF_AH        (OFF_AX1 + AX_BYTES)
#define OFF_Z         (OFF_AH + AH_BYTES)
#define SMEM_BYTES    (OFF_Z + 16 * 128 * 4)   // 312 KB total

// async 16B/lane global->LDS copy (ASYNCcnt-tracked), GVS addressing:
// mem = SADDR(64b sgpr pair) + VADDR(32b byte offset); VDST = LDS byte addr.
#define ASYNC_B128(ldsOff, vOff, sBase)                                  \
    asm volatile("global_load_async_to_lds_b128 %0, %1, %2"              \
                 :: "v"(ldsOff), "v"(vOff), "s"(sBase) : "memory")
#define WAIT_ASYNC(n)                                                    \
    asm volatile("s_wait_asynccnt " #n ::: "memory")

__device__ __forceinline__ float sigm(float x) {
    return 1.0f / (1.0f + __expf(-x));
}
__device__ __forceinline__ float fast_tanh(float x) {
    float e = __expf(-2.0f * fabsf(x));
    float t = (1.0f - e) / (1.0f + e);
    return copysignf(t, x);
}

// x f32 -> bf16 (same [B][T][D] layout), read once, vectorized.
__global__ void cvt_bf16(const float* __restrict__ src, __bf16* __restrict__ dst,
                         int n4) {
    for (int i = blockIdx.x * blockDim.x + threadIdx.x; i < n4;
         i += gridDim.x * blockDim.x) {
        const float4 v = ((const float4*)src)[i];
        __bf16* d = dst + (size_t)i * 4;
        d[0] = (__bf16)v.x; d[1] = (__bf16)v.y;
        d[2] = (__bf16)v.z; d[3] = (__bf16)v.w;
    }
}

// Pack [W;U] (f32, row-major [K,4H]) into per-WG, per-column, K-contiguous
// bf16 blobs matching the WMMA B-fragment read pattern.
__global__ void pack_wu(const float* __restrict__ W, const float* __restrict__ U,
                        __bf16* __restrict__ blob) {
    const int total = NWGN * 128 * 1024;
    for (int idx = blockIdx.x * blockDim.x + threadIdx.x; idx < total;
         idx += gridDim.x * blockDim.x) {
        int K   = idx & 1023;
        int nl  = (idx >> 10) & 127;
        int wgn = idx >> 17;
        int ntile = nl >> 4, n = nl & 15;
        int gate = ntile >> 1, half = ntile & 1;
        int ng = gate * 512 + wgn * 32 + half * 16 + n;
        float v = (K < 512) ? W[K * 2048 + ng] : U[(K - 512) * 2048 + ng];
        blob[idx] = (__bf16)v;
    }
}

template <bool WRITE_SEQ, bool IS_LAST>
__global__ __launch_bounds__(256, 1) void lstm_layer(
    const __bf16* __restrict__ xin, size_t strideM, size_t strideT,
    const __bf16* __restrict__ wblob, const float* __restrict__ bias,
    __bf16* __restrict__ hbuf,          // 2 x [128 x 512] bf16, buf0 zeroed
    __bf16* __restrict__ hseq,          // [T][B][H] bf16 (layer 1 only)
    float* __restrict__ out,            // [B][H] f32 (last layer only)
    unsigned* __restrict__ counters) {  // 8 groups, 256 B apart, zeroed

    extern __shared__ char smem[];
    const unsigned ldsBase = (unsigned)(uintptr_t)smem;  // LDS byte base

    const int tid = threadIdx.x;
    const int wgn = blockIdx.x;   // 0..15 column group
    const int wgm = blockIdx.y;   // 0..7  row group

    // ---- load resident weight slice into LDS (once) ----
    {
        const uint4* src = (const uint4*)(wblob + (size_t)wgn * 131072);
        uint4* dst = (uint4*)smem;
        for (int i = tid; i < LDS_W_BYTES / 16; i += 256) dst[i] = src[i];
    }

    // ---- per-thread async staging addresses (4 x 16B chunks each) ----
    unsigned axLds0[4], axLds1[4], ahLds[4], axOff[4], ahOff[4];
#pragma unroll
    for (int j = 0; j < 4; ++j) {
        const int chunk = tid + 256 * j;        // 0..1023
        const int row   = chunk >> 6;           // 0..15
        const int colb  = (chunk & 63) << 4;    // byte col in 1024B row
        const int cole  = (chunk & 63) << 3;    // element col (bf16)
        const int mg    = wgm * 16 + row;
        axLds0[j] = ldsBase + OFF_AX0 + row * 1024 + colb;
        axLds1[j] = ldsBase + OFF_AX1 + row * 1024 + colb;
        ahLds[j]  = ldsBase + OFF_AH  + row * 1024 + colb;
        axOff[j]  = (unsigned)(((size_t)mg * strideM + cole) * 2);
        ahOff[j]  = (unsigned)((mg * 512 + cole) * 2);
    }

    // ---- per-thread gate-phase constants (2 elements per thread) ----
    const int q  = tid << 1;
    const int gm = q >> 5;        // row within 16-row slice
    const int gc = q & 31;        // even column within 32-col slice
    const int cgl = wgn * 32 + gc;
    const float bi0 = bias[cgl],        bi1 = bias[cgl + 1];
    const float bf0 = bias[512 + cgl],  bf1 = bias[512 + cgl + 1];
    const float bg0 = bias[1024 + cgl], bg1 = bias[1024 + cgl + 1];
    const float bo0 = bias[1536 + cgl], bo1 = bias[1536 + cgl + 1];
    float c0v = 0.0f, c1v = 0.0f;

    // ---- WMMA lane constants ----
    const int w    = tid >> 5;          // wave = N tile (0..7)
    const int lane = tid & 31;
    const int l15  = lane & 15;
    const int kh   = lane >> 4;
    const char* bB  = (const char*)smem + (w * 16 + l15) * 2048 + kh * 32;
    const int   aRow = l15 * 1024 + kh * 32;
    float* ldsZ = (float*)(smem + OFF_Z);

    unsigned* cntp = counters + wgm * 64;

    // ---- prime: async-load x-tile for t = 0 into buffer 0 ----
#pragma unroll
    for (int j = 0; j < 4; ++j) ASYNC_B128(axLds0[j], axOff[j], xin);

    for (int t = 0; t < Tsz; ++t) {
        const __bf16* hprev = hbuf + ((t & 1) ? 65536 : 0);
        __bf16*       hnext = hbuf + ((t & 1) ? 0 : 65536);

        // ---- async-load h-half for this step (hprev ready after barrier) --
#pragma unroll
        for (int j = 0; j < 4; ++j) ASYNC_B128(ahLds[j], ahOff[j], hprev);

        // ---- async-prefetch next step's x-half into the other buffer ----
        {
            const int tn = (t + 1) & (Tsz - 1);     // wraps harmlessly at end
            const __bf16* xn = xin + (size_t)tn * strideT;
            if (t & 1) {
#pragma unroll
                for (int j = 0; j < 4; ++j) ASYNC_B128(axLds0[j], axOff[j], xn);
            } else {
#pragma unroll
                for (int j = 0; j < 4; ++j) ASYNC_B128(axLds1[j], axOff[j], xn);
            }
        }

        // wait for this step's x-tile + h-tile; leave the 4 prefetch loads
        // (issued last) in flight.
        WAIT_ASYNC(4);
        __syncthreads();

        // ---- z = [x|h] @ [W;U] for this 16x16 tile, K = 1024 ----
        const char* aXb = (const char*)smem + ((t & 1) ? OFF_AX1 : OFF_AX0) + aRow;
        const char* aHb = (const char*)smem + OFF_AH + aRow;
        v8f acc0 = {}, acc1 = {};
#pragma unroll
        for (int ks = 0; ks < 16; ++ks) {
            v8bf alo = *(const v8bf*)(aXb + ks * 64);
            v8bf ahi = *(const v8bf*)(aXb + ks * 64 + 16);
            v16bf a = __builtin_shufflevector(alo, ahi, 0, 1, 2, 3, 4, 5, 6, 7,
                                              8, 9, 10, 11, 12, 13, 14, 15);
            v16bf b = *(const v16bf*)(bB + ks * 64);
            if (ks & 1)
                acc1 = __builtin_amdgcn_wmma_f32_16x16x32_bf16(
                    false, a, false, b, (short)0, acc1, false, false);
            else
                acc0 = __builtin_amdgcn_wmma_f32_16x16x32_bf16(
                    false, a, false, b, (short)0, acc0, false, false);
        }
#pragma unroll
        for (int ks = 0; ks < 16; ++ks) {
            v8bf alo = *(const v8bf*)(aHb + ks * 64);
            v8bf ahi = *(const v8bf*)(aHb + ks * 64 + 16);
            v16bf a = __builtin_shufflevector(alo, ahi, 0, 1, 2, 3, 4, 5, 6, 7,
                                              8, 9, 10, 11, 12, 13, 14, 15);
            v16bf b = *(const v16bf*)(bB + (16 + ks) * 64);
            if (ks & 1)
                acc1 = __builtin_amdgcn_wmma_f32_16x16x32_bf16(
                    false, a, false, b, (short)0, acc1, false, false);
            else
                acc0 = __builtin_amdgcn_wmma_f32_16x16x32_bf16(
                    false, a, false, b, (short)0, acc0, false, false);
        }
        const v8f acc = acc0 + acc1;
#pragma unroll
        for (int r = 0; r < 8; ++r)
            ldsZ[(kh * 8 + r) * 128 + w * 16 + l15] = acc[r];
        __syncthreads();

        // ---- gates: i,f,g,o -> c,h update (c in registers) ----
        {
            const float* zr = ldsZ + gm * 128;
            float i0 = sigm(zr[gc] + bi0),            i1 = sigm(zr[gc + 1] + bi1);
            float f0 = sigm(zr[32 + gc] + bf0),       f1 = sigm(zr[33 + gc] + bf1);
            float g0 = fast_tanh(zr[64 + gc] + bg0),  g1 = fast_tanh(zr[65 + gc] + bg1);
            float o0 = sigm(zr[96 + gc] + bo0),       o1 = sigm(zr[97 + gc] + bo1);
            c0v = f0 * c0v + i0 * g0;
            c1v = f1 * c1v + i1 * g1;
            float h0 = o0 * fast_tanh(c0v);
            float h1 = o1 * fast_tanh(c1v);
            const int mg = wgm * 16 + gm;
            const size_t hoff = (size_t)mg * 512 + cgl;
            hnext[hoff]     = (__bf16)h0;
            hnext[hoff + 1] = (__bf16)h1;
            if (WRITE_SEQ) {
                __bf16* sd = hseq + (size_t)t * (Bsz * Hsz) + hoff;
                sd[0] = (__bf16)h0;
                sd[1] = (__bf16)h1;
            }
            if (IS_LAST && t == Tsz - 1) {
                out[hoff]     = h0;
                out[hoff + 1] = h1;
            }
        }

        // ---- barrier among the 16 WGs sharing this row group ----
        if (t != Tsz - 1) {
            __syncthreads();
            if (tid == 0) {
                __threadfence();
                atomicAdd(cntp, 1u);
                const unsigned tgt = (unsigned)(NWGN) * (unsigned)(t + 1);
                while (__hip_atomic_load(cntp, __ATOMIC_ACQUIRE,
                                         __HIP_MEMORY_SCOPE_AGENT) < tgt) {
                    __builtin_amdgcn_s_sleep(2);
                }
                __threadfence();
            }
            __syncthreads();
        }
    }
}

extern "C" void kernel_launch(void* const* d_in, const int* in_sizes, int n_in,
                              void* d_out, int out_size, void* d_ws, size_t ws_size,
                              hipStream_t stream) {
    (void)in_sizes; (void)n_in; (void)out_size; (void)ws_size;

    const float* x  = (const float*)d_in[0];   // [128,512,512]
    const float* W1 = (const float*)d_in[1];   // [512,2048]
    const float* U1 = (const float*)d_in[2];   // [512,2048]
    const float* b1 = (const float*)d_in[3];   // [2048]
    const float* W2 = (const float*)d_in[4];
    const float* U2 = (const float*)d_in[5];
    const float* b2 = (const float*)d_in[6];
    float* out = (float*)d_out;                // [128,512]

    // workspace layout
    char* ws = (char*)d_ws;
    const size_t WB_BYTES  = (size_t)NWGN * 128 * 1024 * 2;          // 4 MiB each
    const size_t SEQ_BYTES = (size_t)Tsz * Bsz * Hsz * 2;            // 32 M elems
    const size_t XBF_BYTES = (size_t)Bsz * Tsz * 512 * 2;
    __bf16* wb1   = (__bf16*)(ws);
    __bf16* wb2   = (__bf16*)(ws + WB_BYTES);
    __bf16* h1seq = (__bf16*)(ws + 2 * WB_BYTES);
    __bf16* xbf   = (__bf16*)(ws + 2 * WB_BYTES + SEQ_BYTES);
    __bf16* hbuf  = (__bf16*)(ws + 2 * WB_BYTES + SEQ_BYTES + XBF_BYTES);
    unsigned* cnt = (unsigned*)(ws + 2 * WB_BYTES + SEQ_BYTES + XBF_BYTES + 262144);

    // allow 312 KB dynamic LDS on the persistent kernels
    (void)hipFuncSetAttribute((const void*)lstm_layer<true, false>,
                              hipFuncAttributeMaxDynamicSharedMemorySize, SMEM_BYTES);
    (void)hipFuncSetAttribute((const void*)lstm_layer<false, true>,
                              hipFuncAttributeMaxDynamicSharedMemorySize, SMEM_BYTES);

    // prologue: x -> bf16 once; pack weights into WMMA-B-friendly blobs
    cvt_bf16<<<dim3(4096), dim3(256), 0, stream>>>(
        x, xbf, (int)(Bsz * Tsz * 512 / 4));
    pack_wu<<<dim3(2048), dim3(256), 0, stream>>>(W1, U1, wb1);
    pack_wu<<<dim3(2048), dim3(256), 0, stream>>>(W2, U2, wb2);

    const dim3 grid(NWGN, NWGM), block(256);

    // ---- layer 1: input x_bf16 [B][T][D], write full h sequence ----
    (void)hipMemsetAsync(hbuf, 0, 131072, stream);  // h0 = 0 (ping buffer)
    (void)hipMemsetAsync(cnt, 0, 2048, stream);     // barrier counters
    lstm_layer<true, false><<<grid, block, SMEM_BYTES, stream>>>(
        xbf, (size_t)(Tsz * 512), (size_t)512,
        wb1, b1, hbuf, h1seq, nullptr, cnt);

    // ---- layer 2: input h1 sequence (bf16, [T][B][H]), final h -> out ----
    (void)hipMemsetAsync(hbuf, 0, 131072, stream);
    (void)hipMemsetAsync(cnt, 0, 2048, stream);
    lstm_layer<false, true><<<grid, block, SMEM_BYTES, stream>>>(
        h1seq, (size_t)512, (size_t)(Bsz * Hsz),
        wb2, b2, hbuf, nullptr, out, cnt);
}
